// SelfAttentionBlock_1649267442305
// MI455X (gfx1250) — matrix operand order repaired and checked by reading, not compiled, not run on hardware
//
#include <hip/hip_runtime.h>
#include <hip/hip_bf16.h>

#define BB 4
#define SS 2048
#define EE 1024
#define HH 16
#define DD 64

typedef __attribute__((ext_vector_type(16))) __bf16 v16bf;
typedef __attribute__((ext_vector_type(8)))  __bf16 v8bf;
typedef __attribute__((ext_vector_type(8)))  float  v8f;
typedef __attribute__((ext_vector_type(4)))  float  f4;
typedef __attribute__((ext_vector_type(4)))  unsigned int v4u;
typedef __attribute__((ext_vector_type(8)))  int v8i;
typedef __attribute__((ext_vector_type(4)))  int v4i;

#if defined(__gfx1250__) && __has_builtin(__builtin_amdgcn_tensor_load_to_lds)
#define USE_TDM 1
#else
#define USE_TDM 0
#endif

// ---------------------------------------------------------------------------
// TDM: DMA a 2D tile (tile_d0 x tile_d1, 2-byte elements) from global memory
// into LDS. Descriptor per CDNA5 ISA (D# groups 0/1,
// groups 2/3 zero for a 2D tensor). Tracked by TENSORcnt.
// ---------------------------------------------------------------------------
#if USE_TDM
__device__ __forceinline__ void tdm_load_2d(
    unsigned int lds_addr, const void* gaddr,
    unsigned int tensor_d0, unsigned int tensor_d1, unsigned int stride0,
    unsigned int tile_d0, unsigned int tile_d1)
{
    unsigned long long ga = (unsigned long long)gaddr;
    v4u g0;
    g0[0] = 1u;                                       // count=1 (valid user D#)
    g0[1] = lds_addr;                                 // lds_addr (bytes)
    g0[2] = (unsigned int)ga;                         // global_addr[31:0]
    g0[3] = (unsigned int)((ga >> 32) & 0x01FFFFFFull)// global_addr[56:32]
          | 0x80000000u;                              // type=2 ("image")
    v8i g1;
    g1[0] = 0x00010000;                               // data_size=1 (2 bytes)
    g1[1] = (int)(tensor_d0 << 16);                   // tensor_dim0[15:0] @ b48
    g1[2] = (int)((tensor_d0 >> 16) | (tensor_d1 << 16));
    g1[3] = (int)((tensor_d1 >> 16) | (tile_d0 << 16)); // tile_dim0 @ b112
    g1[4] = (int)tile_d1;                             // tile_dim1 @ b128
    g1[5] = (int)stride0;                             // dim0_stride[31:0]
    g1[6] = 0;
    g1[7] = 0;
    v4i z4 = {0, 0, 0, 0};
#if __has_include(<hip/amd_detail/amd_gfx1250_TDM.h>)
    v8i z8 = {0, 0, 0, 0, 0, 0, 0, 0};
    __builtin_amdgcn_tensor_load_to_lds(g0, g1, z4, z4, z8, 0);
#else
    __builtin_amdgcn_tensor_load_to_lds(g0, g1, z4, z4, 0);
#endif
}

__device__ __forceinline__ void tdm_wait()
{
#if __has_builtin(__builtin_amdgcn_s_wait_tensorcnt)
    __builtin_amdgcn_s_wait_tensorcnt(0);
#else
    asm volatile("s_wait_tensorcnt 0x0" ::: "memory");
#endif
}
#endif

// ---------------------------------------------------------------------------
// Stage 0: QKV projection. One wave per 16-row tile of one head.
// WMMA: D(16x16 f32) = A(16x32 bf16 x-rows) x B(32x16 bf16 W^T) + C, 2 chunks.
// Q,K -> [b,h,s,d] bf16 ; V -> transposed [b,h,d,s] bf16.
// ---------------------------------------------------------------------------
__global__ __launch_bounds__(128) void qkv_proj_kernel(
    const float* __restrict__ x, const float* __restrict__ Wq,
    const float* __restrict__ Wk, const float* __restrict__ Wv,
    __bf16* __restrict__ Qb, __bf16* __restrict__ Kb, __bf16* __restrict__ Vt)
{
    const int tid  = threadIdx.x;
    const int lane = tid & 31;
    const int ln   = lane & 15;
    const int hi   = lane >> 4;
    const int wave = tid >> 5;

    const int tile = blockIdx.x * 4 + wave;      // B*H*(S/16) = 8192 tiles
    const int st = tile & 127;
    const int h  = (tile >> 7) & 15;
    const int b  = tile >> 11;
    const int s0 = st * 16;

    const float* xrow = x + ((size_t)(b * SS + s0 + ln) * EE + h * DD);
    v16bf A[2];
    for (int c = 0; c < 2; ++c) {
        const int k0 = hi * 8 + 32 * c;
        f4 a0 = *(const f4*)(xrow + k0);
        f4 a1 = *(const f4*)(xrow + k0 + 4);
        f4 c0 = *(const f4*)(xrow + k0 + 16);
        f4 c1 = *(const f4*)(xrow + k0 + 20);
        for (int i = 0; i < 4; ++i) {
            A[c][i]      = (__bf16)a0[i];
            A[c][i + 4]  = (__bf16)a1[i];
            A[c][i + 8]  = (__bf16)c0[i];
            A[c][i + 12] = (__bf16)c1[i];
        }
    }

    const float* Ws[3] = {Wq, Wk, Wv};
    const size_t headBase = ((size_t)(b * HH + h)) * SS * DD;
    const size_t vtBase   = ((size_t)(b * HH + h)) * DD * SS;

    for (int w = 0; w < 3; ++w) {
        const float* W = Ws[w];
        for (int ot = 0; ot < 4; ++ot) {
            const int o = ot * 16 + ln;
            v8f acc = {};
            for (int c = 0; c < 2; ++c) {
                const int d0 = hi * 8 + 32 * c;
                const float* wr = W + o * DD + d0;
                f4 w0 = *(const f4*)(wr);
                f4 w1 = *(const f4*)(wr + 4);
                f4 w2 = *(const f4*)(wr + 16);
                f4 w3 = *(const f4*)(wr + 20);
                v16bf Bf;
                for (int i = 0; i < 4; ++i) {
                    Bf[i]      = (__bf16)w0[i];
                    Bf[i + 4]  = (__bf16)w1[i];
                    Bf[i + 8]  = (__bf16)w2[i];
                    Bf[i + 12] = (__bf16)w3[i];
                }
                acc = __builtin_amdgcn_wmma_f32_16x16x32_bf16(
                        false, A[c], false, Bf, (short)0, acc, false, false);
            }
            if (w < 2) {
                __bf16* dst = (w == 0 ? Qb : Kb) + headBase;
                for (int r = 0; r < 8; ++r)
                    dst[(size_t)(s0 + r + hi * 8) * DD + o] = (__bf16)acc[r];
            } else {
                v8bf pk;
                for (int r = 0; r < 8; ++r) pk[r] = (__bf16)acc[r];
                *(v8bf*)(Vt + vtBase + (size_t)o * SS + s0 + hi * 8) = pk;
            }
        }
    }
}

// ---------------------------------------------------------------------------
// Stage 1: fused causal flash attention (transposed-score formulation):
//   S^T[key,q] = sum_d K[key,d]*Q[q,d]          (A = K tile, B = Q frag)
//   out^T[d,q] = sum_key V^T[d,key]*P^T[key,q]  (A = V^T tile, B = P frag)
// The D-layout of S^T becomes the B-layout of P via one shfl_xor(16) swap,
// so P never leaves registers. K/V^T tiles delivered to LDS by the TDM.
// ---------------------------------------------------------------------------
__global__ __launch_bounds__(128) void flash_attn_kernel(
    const __bf16* __restrict__ Qb, const __bf16* __restrict__ Kb,
    const __bf16* __restrict__ Vt, float* __restrict__ out)
{
    __shared__ __attribute__((aligned(16))) __bf16 Ksh[32 * DD]; // [key][d]
    __shared__ __attribute__((aligned(16))) __bf16 Vsh[DD * 32]; // [d][key]

    const int tid  = threadIdx.x;
    const int lane = tid & 31;
    const int ln   = lane & 15;
    const int hi   = lane >> 4;
    const int wave = tid >> 5;

    const int blk = blockIdx.x;              // B*H*(S/64) = 2048
    const int qb  = (blk & 31) * 64;
    const int h   = (blk >> 5) & 15;
    const int b   = blk >> 9;

    const int q0 = qb + wave * 16;
    const int qg = q0 + ln;

    const __bf16* Qh = Qb + ((size_t)(b * HH + h)) * SS * DD;
    const __bf16* Kh = Kb + ((size_t)(b * HH + h)) * SS * DD;
    const __bf16* Vh = Vt + ((size_t)(b * HH + h)) * DD * SS;

    v16bf Qf[2];
    for (int c = 0; c < 2; ++c)
        Qf[c] = *(const v16bf*)(Qh + (size_t)(q0 + ln) * DD + hi * 16 + 32 * c);

    float m = -INFINITY, l = 0.f;
    v8f acc[4];
    for (int mt = 0; mt < 4; ++mt) acc[mt] = v8f{};

    const float scale = 0.125f;
    const int kend_blk = qb + 64;
    const int kend_w   = q0 + 16;

#if USE_TDM
    const unsigned int ldsK = (unsigned int)(size_t)(void*)Ksh;
    const unsigned int ldsV = (unsigned int)(size_t)(void*)Vsh;
#endif

    for (int kt = 0; kt < kend_blk; kt += 32) {
#if USE_TDM
        // Tensor Data Mover: DMA K tile (32 keys x 64 d) and V^T tile
        // (64 d x 32 keys) into LDS; one issue per block (EXEC ignored).
        if (wave == 0) {
            tdm_load_2d(ldsK, Kh + (size_t)kt * DD, DD, SS, DD, DD, 32);
            tdm_load_2d(ldsV, Vh + kt,              SS, DD, SS, 32, DD);
            tdm_wait();
        }
#else
        {
            const uint4* kg4 = (const uint4*)(Kh + (size_t)kt * DD);
            uint4* ks4 = (uint4*)Ksh;
            ks4[tid]       = kg4[tid];
            ks4[tid + 128] = kg4[tid + 128];
            uint4* vs4 = (uint4*)Vsh;
            const int u0 = tid, u1 = tid + 128;
            vs4[u0] = *(const uint4*)(Vh + (size_t)(u0 >> 2) * SS + kt + (u0 & 3) * 8);
            vs4[u1] = *(const uint4*)(Vh + (size_t)(u1 >> 2) * SS + kt + (u1 & 3) * 8);
        }
#endif
        __syncthreads();

        if (kt < kend_w) {                   // wave-uniform -> EXEC all 1s
            v8f st[2];
            for (int t = 0; t < 2; ++t) {
                v8f s = {};
                for (int c = 0; c < 2; ++c) {
                    const int koff = hi * 16 + 32 * c;
                    const __bf16* kr = Ksh + (t * 16 + ln) * DD;
                    v8bf lo = *(const v8bf*)(kr + koff);
                    v8bf up = *(const v8bf*)(kr + koff + 8);
                    v16bf Af = __builtin_shufflevector(lo, up,
                        0, 1, 2, 3, 4, 5, 6, 7, 8, 9, 10, 11, 12, 13, 14, 15);
                    s = __builtin_amdgcn_wmma_f32_16x16x32_bf16(
                            false, Af, false, Qf[c], (short)0, s, false, false);
                }
                st[t] = s;
            }
            float tmax = -INFINITY;
            for (int t = 0; t < 2; ++t)
                for (int r = 0; r < 8; ++r) {
                    const int kidx = kt + t * 16 + r + hi * 8;
                    float sv = st[t][r];
                    if (kidx > qg) sv = -INFINITY;
                    st[t][r] = sv;
                    tmax = fmaxf(tmax, sv);
                }
            tmax = fmaxf(tmax, __shfl_xor(tmax, 16, 32));
            const float mnew  = fmaxf(m, tmax);
            const float alpha = __expf((m - mnew) * scale);
            float p0[8], p1[8], rsum = 0.f;
            for (int r = 0; r < 8; ++r) {
                p0[r] = __expf((st[0][r] - mnew) * scale);
                p1[r] = __expf((st[1][r] - mnew) * scale);
                rsum += p0[r] + p1[r];
            }
            rsum += __shfl_xor(rsum, 16, 32);
            l = l * alpha + rsum;
            m = mnew;

            v16bf Pb;
            for (int r = 0; r < 8; ++r) {
                const float a0 = __shfl_xor(p0[r], 16, 32);
                const float a1 = __shfl_xor(p1[r], 16, 32);
                Pb[r]     = (__bf16)(hi ? a1    : p0[r]);
                Pb[r + 8] = (__bf16)(hi ? p1[r] : a0);
            }
            for (int mt = 0; mt < 4; ++mt) {
                for (int r = 0; r < 8; ++r) acc[mt][r] *= alpha;
                const int koff = hi * 16;
                const __bf16* vr = Vsh + (mt * 16 + ln) * 32;
                v8bf lo = *(const v8bf*)(vr + koff);
                v8bf up = *(const v8bf*)(vr + koff + 8);
                v16bf Af = __builtin_shufflevector(lo, up,
                    0, 1, 2, 3, 4, 5, 6, 7, 8, 9, 10, 11, 12, 13, 14, 15);
                acc[mt] = __builtin_amdgcn_wmma_f32_16x16x32_bf16(
                            false, Af, false, Pb, (short)0, acc[mt], false, false);
            }
        }
        __syncthreads();
    }

    const float inv = 1.f / l;
    for (int mt = 0; mt < 4; ++mt) {
        f4 o0, o1;
        for (int r = 0; r < 4; ++r) {
            o0[r] = acc[mt][r]     * inv;
            o1[r] = acc[mt][r + 4] * inv;
        }
        float* dst = out + ((size_t)(b * SS + q0 + ln) * EE + h * DD + mt * 16 + hi * 8);
        *(f4*)dst       = o0;
        *(f4*)(dst + 4) = o1;
    }
}

extern "C" void kernel_launch(void* const* d_in, const int* in_sizes, int n_in,
                              void* d_out, int out_size, void* d_ws, size_t ws_size,
                              hipStream_t stream) {
    (void)in_sizes; (void)n_in; (void)out_size; (void)ws_size;
    const float* x  = (const float*)d_in[0];
    const float* Wq = (const float*)d_in[1];
    const float* Wk = (const float*)d_in[2];
    const float* Wv = (const float*)d_in[3];

    const size_t perTensor = (size_t)BB * HH * SS * DD;   // 8M bf16 elements
    __bf16* Qb = (__bf16*)d_ws;
    __bf16* Kb = Qb + perTensor;
    __bf16* Vt = Kb + perTensor;                          // 48 MB of ws

    qkv_proj_kernel<<<2048, 128, 0, stream>>>(x, Wq, Wk, Wv, Qb, Kb, Vt);
    flash_attn_kernel<<<2048, 128, 0, stream>>>(Qb, Kb, Vt, (float*)d_out);
}
